// GAT_76596446757095
// MI455X (gfx1250) — matrix operand order repaired and checked by reading, not compiled, not run on hardware
//
#include <hip/hip_runtime.h>
#include <hip/hip_bf16.h>

// ---------------------------------------------------------------------------
// GAT 2-layer forward for MI455X (gfx1250, wave32, WMMA).
// ---------------------------------------------------------------------------

typedef __attribute__((ext_vector_type(16))) _Float16 v16h;
typedef __attribute__((ext_vector_type(8)))  _Float16 v8h;
typedef __attribute__((ext_vector_type(8)))  float    v8f;

#define NEG_SLOPE 0.2f

// ---- monotonic float<->uint mapping for atomicMax on floats ---------------
__device__ __forceinline__ unsigned f2o(float f) {
  unsigned u = __float_as_uint(f);
  return (u & 0x80000000u) ? ~u : (u | 0x80000000u);
}
__device__ __forceinline__ float o2f(unsigned u) {
  return (u & 0x80000000u) ? __uint_as_float(u & 0x7fffffffu)
                           : __uint_as_float(~u);
}
#define NEG_INF_ORD 0x007FFFFFu   // f2o(-inf)

// ---------------------------------------------------------------------------
// Utility kernels
// ---------------------------------------------------------------------------
__global__ void gat_fill_u32(unsigned* __restrict__ p, unsigned v, long n) {
  long i = (long)blockIdx.x * blockDim.x + threadIdx.x;
  if (i < n) p[i] = v;
}

__global__ void gat_cvt_f16(const float* __restrict__ s, _Float16* __restrict__ d, long n) {
  long i = (long)blockIdx.x * blockDim.x + threadIdx.x;
  if (i < n) d[i] = (_Float16)s[i];
}

// W1 [256,256] row-major f32 -> transposed f16  Wt1[col][k] (k contiguous)
__global__ void gat_cvt_w1t(const float* __restrict__ s, _Float16* __restrict__ d) {
  int i = blockIdx.x * blockDim.x + threadIdx.x;   // over 256*256
  if (i >= 256 * 256) return;
  int c = i >> 8, k = i & 255;                     // d index: [c][k]
  d[i] = (_Float16)s[k * 256 + c];
}

// W2 [256,40] row-major f32 -> transposed f16 padded  Wt2[48][256] (cols 40..47 = 0)
__global__ void gat_cvt_w2t(const float* __restrict__ s, _Float16* __restrict__ d) {
  int i = blockIdx.x * blockDim.x + threadIdx.x;   // over 48*256
  if (i >= 48 * 256) return;
  int c = i >> 8, k = i & 255;
  d[i] = (c < 40) ? (_Float16)s[k * 40 + c] : (_Float16)0.0f;
}

// ---------------------------------------------------------------------------
// WMMA GEMM:  D[nrows,outc] = A[nrows,256]f16 * Bt[outc,256]f16^T   (K = 256)
// Bt is the pre-transposed weight: Bt[col][k], k contiguous -> all staging is
// 16B vectorized (global_load_b128 -> ds_store_b128).
// Block: 256 threads = 8 waves. Block tile: 128 rows x (NT*16) cols.
// Each wave owns one 16-row subtile and NT col tiles; K staged in 32-slabs
// through LDS. Fragments are two contiguous 16B LDS loads per operand,
// matching the documented 16-bit A/B lane layouts.
// ---------------------------------------------------------------------------
template<int NT>
__global__ __launch_bounds__(256)
void gat_wmma_gemm(const _Float16* __restrict__ A, const _Float16* __restrict__ Bt,
                   float* __restrict__ D, int nrows, int lda, int ldd, int outc) {
  constexpr int BK = 32;
  constexpr int LW = BK + 8;                 // 40 halves: rows stay 16B aligned
  __shared__ _Float16 Ash[128][LW];
  __shared__ _Float16 Bsh[NT * 16][LW];

  const int tid  = threadIdx.x;
  const int lane = tid & 31;
  const int wave = tid >> 5;                 // 0..7
  const int row0 = blockIdx.x * 128;
  const int col0 = blockIdx.y * (NT * 16);

  v8f acc[NT] = {};

  for (int k0 = 0; k0 < 256; k0 += BK) {
    // ---- stage A: 128 x 32 halves as 512 16B chunks, 2 per thread --------
    #pragma unroll
    for (int it = 0; it < 2; ++it) {
      int idx  = tid + 256 * it;             // 0..511
      int r    = idx >> 2;
      int c8   = idx & 3;
      int grow = row0 + r;
      if (grow >= nrows) grow = nrows - 1;   // clamp (stores are guarded)
      *(v8h*)&Ash[r][c8 * 8] = *(const v8h*)&A[(long)grow * lda + k0 + c8 * 8];
    }
    // ---- stage B (already transposed): NT*16 cols x 32 halves, 16B chunks
    for (int idx = tid; idx < NT * 16 * 4; idx += 256) {
      int c  = idx >> 2;
      int c8 = idx & 3;
      *(v8h*)&Bsh[c][c8 * 8] = *(const v8h*)&Bt[(long)(col0 + c) * 256 + k0 + c8 * 8];
    }
    if (k0 + BK < 256)                       // hint next A slab into near caches
      __builtin_prefetch(&A[(long)(row0 + (tid >> 1)) * lda + k0 + BK], 0, 3);
    __syncthreads();

    // ---- A fragment: lane<16 -> K 0..7 & 16..23 ; lane>=16 -> 8..15 & 24..31
    const int arow = wave * 16 + (lane & 15);
    const int hs   = lane >> 4;
    v8h alo = *(const v8h*)&Ash[arow][hs * 8];
    v8h ahi = *(const v8h*)&Ash[arow][16 + hs * 8];
    v16h a;
    #pragma unroll
    for (int i = 0; i < 8; ++i) { a[i] = alo[i]; a[8 + i] = ahi[i]; }

    #pragma unroll
    for (int t = 0; t < NT; ++t) {
      // B fragment: lane -> column (lane&15), K (lane>>4)*8..+7 & 16+(lane>>4)*8..+7
      const _Float16* bp = &Bsh[t * 16 + (lane & 15)][hs * 8];
      v8h blo = *(const v8h*)bp;
      v8h bhi = *(const v8h*)(bp + 16);
      v16h b;
      #pragma unroll
      for (int i = 0; i < 8; ++i) { b[i] = blo[i]; b[8 + i] = bhi[i]; }
      acc[t] = __builtin_amdgcn_wmma_f32_16x16x32_f16(
          false, a, false, b, (short)0, acc[t], false, false);
    }
    __syncthreads();
  }

  // ---- store: VGPR j holds D[wave*16 + j + 8*(lane>>4)][lane&15] ---------
  #pragma unroll
  for (int t = 0; t < NT; ++t) {
    int cc = col0 + t * 16 + (lane & 15);
    if (cc < outc) {
      #pragma unroll
      for (int j = 0; j < 8; ++j) {
        int rr = row0 + wave * 16 + j + 8 * (lane >> 4);
        if (rr < nrows) D[(long)rr * ldd + cc] = acc[t][j];
      }
    }
  }
}

// ---------------------------------------------------------------------------
// alpha_{src,dst} dot products
// ---------------------------------------------------------------------------
__global__ void gat_alpha_l1(const float* __restrict__ h,
                             const float* __restrict__ as_w, const float* __restrict__ ad_w,
                             float* __restrict__ as, float* __restrict__ ad, int n) {
  long t = (long)blockIdx.x * blockDim.x + threadIdx.x;
  if (t >= (long)n * 8) return;
  int node = (int)(t >> 3), hh = (int)(t & 7);
  const float* hp = h + (long)node * 256 + hh * 32;
  const float* sp = as_w + hh * 32;
  const float* dp = ad_w + hh * 32;
  float ss = 0.f, sd = 0.f;
  #pragma unroll
  for (int c = 0; c < 32; ++c) { ss += hp[c] * sp[c]; sd += hp[c] * dp[c]; }
  as[t] = ss; ad[t] = sd;
}

__global__ void gat_alpha_l2(const float* __restrict__ g,   // ld 48
                             const float* __restrict__ as_w, const float* __restrict__ ad_w,
                             float* __restrict__ as, float* __restrict__ ad, int n) {
  long t = (long)blockIdx.x * blockDim.x + threadIdx.x;
  if (t >= n) return;
  const float* gp = g + t * 48;
  float ss = 0.f, sd = 0.f;
  #pragma unroll
  for (int c = 0; c < 40; ++c) { ss += gp[c] * as_w[c]; sd += gp[c] * ad_w[c]; }
  as[t] = ss; ad[t] = sd;
}

// ---------------------------------------------------------------------------
// Edge softmax passes (H = heads). Thread per (edge, head).
// Edges e < E come from edge_index; e >= E are self loops (node e-E).
// ---------------------------------------------------------------------------
template<int H>
__global__ void gat_edge_max(const int* __restrict__ src, const int* __restrict__ dst,
                             int E, int Nn, const float* __restrict__ as,
                             const float* __restrict__ ad, unsigned* __restrict__ m) {
  long t = (long)blockIdx.x * blockDim.x + threadIdx.x;
  long total = (long)(E + Nn) * H;
  if (t >= total) return;
  long e = t / H; int hh = (int)(t % H);
  int s, d;
  if (e < E) { s = src[e]; d = dst[e]; } else { s = d = (int)(e - E); }
  float v = as[(long)s * H + hh] + ad[(long)d * H + hh];
  v = (v >= 0.f) ? v : NEG_SLOPE * v;
  atomicMax(&m[(long)d * H + hh], f2o(v));
}

template<int H>
__global__ void gat_edge_sum(const int* __restrict__ src, const int* __restrict__ dst,
                             int E, int Nn, const float* __restrict__ as,
                             const float* __restrict__ ad, const unsigned* __restrict__ m,
                             float* __restrict__ denom) {
  long t = (long)blockIdx.x * blockDim.x + threadIdx.x;
  long total = (long)(E + Nn) * H;
  if (t >= total) return;
  long e = t / H; int hh = (int)(t % H);
  int s, d;
  if (e < E) { s = src[e]; d = dst[e]; } else { s = d = (int)(e - E); }
  float v = as[(long)s * H + hh] + ad[(long)d * H + hh];
  v = (v >= 0.f) ? v : NEG_SLOPE * v;
  float ex = __expf(v - o2f(m[(long)d * H + hh]));
  atomicAdd(&denom[(long)d * H + hh], ex);
}

// Layer-1 aggregation: one wave32 per edge, 8 heads x 32 channels.
// Lanes 0..7 compute per-head weights once; broadcast via shfl; then 8
// coalesced 32-lane gather + atomic-add sweeps over the 256 features.
__global__ __launch_bounds__(256)
void gat_edge_agg_h8(const int* __restrict__ src, const int* __restrict__ dst,
                     int E, int Nn, const float* __restrict__ as,
                     const float* __restrict__ ad, const unsigned* __restrict__ m,
                     const float* __restrict__ denom, const float* __restrict__ hfeat,
                     float* __restrict__ agg) {
  int  lane = threadIdx.x & 31;
  long wv   = (long)blockIdx.x * (blockDim.x >> 5) + (threadIdx.x >> 5);
  long tot  = (long)E + Nn;
  if (wv >= tot) return;
  int s, d;
  if (wv < E) { s = src[wv]; d = dst[wv]; } else { s = d = (int)(wv - E); }
  float w = 0.f;
  if (lane < 8) {
    long ix = (long)d * 8 + lane;
    float v = as[(long)s * 8 + lane] + ad[ix];
    v = (v >= 0.f) ? v : NEG_SLOPE * v;
    float ex = __expf(v - o2f(m[ix]));
    w = ex / (denom[ix] + 1e-16f);
  }
  const float* hs = hfeat + (long)s * 256;
  float*       ag = agg   + (long)d * 256;
  #pragma unroll
  for (int i = 0; i < 8; ++i) {
    float wi = __shfl(w, i, 32);
    atomicAdd(&ag[i * 32 + lane], hs[i * 32 + lane] * wi);
  }
}

// Layer-2 aggregation: one wave32 per edge, 1 head x 40 channels (g ld 48).
__global__ __launch_bounds__(256)
void gat_edge_agg_h1(const int* __restrict__ src, const int* __restrict__ dst,
                     int E, int Nn, const float* __restrict__ as,
                     const float* __restrict__ ad, const unsigned* __restrict__ m,
                     const float* __restrict__ denom, const float* __restrict__ g,
                     float* __restrict__ agg) {
  int  lane = threadIdx.x & 31;
  long wv   = (long)blockIdx.x * (blockDim.x >> 5) + (threadIdx.x >> 5);
  long tot  = (long)E + Nn;
  if (wv >= tot) return;
  int s, d;
  if (wv < E) { s = src[wv]; d = dst[wv]; } else { s = d = (int)(wv - E); }
  float v = as[s] + ad[d];
  v = (v >= 0.f) ? v : NEG_SLOPE * v;
  float w = __expf(v - o2f(m[d])) / (denom[d] + 1e-16f);
  const float* gs = g   + (long)s * 48;
  float*       ag = agg + (long)d * 40;
  for (int c = lane; c < 40; c += 32) atomicAdd(&ag[c], gs[c] * w);
}

// h2 = elu(agg1 + bias1), recast to f16 for the layer-2 GEMM.
__global__ void gat_elu_f16(const float* __restrict__ agg, const float* __restrict__ bias,
                            _Float16* __restrict__ h2, long n) {
  long i = (long)blockIdx.x * blockDim.x + threadIdx.x;
  if (i >= n) return;
  float v = agg[i] + bias[i & 255];
  v = (v > 0.f) ? v : (__expf(v) - 1.f);
  h2[i] = (_Float16)v;
}

__global__ void gat_final_bias(const float* __restrict__ agg, const float* __restrict__ bias,
                               float* __restrict__ out, long n) {
  long i = (long)blockIdx.x * blockDim.x + threadIdx.x;
  if (i >= n) return;
  out[i] = agg[i] + bias[(int)(i % 40)];
}

// ---------------------------------------------------------------------------
// Launcher
// ---------------------------------------------------------------------------
static inline char* ws_alloc(char*& p, size_t bytes) {
  char* r = p;
  p += (bytes + 255) & ~(size_t)255;
  return r;
}

extern "C" void kernel_launch(void* const* d_in, const int* in_sizes, int n_in,
                              void* d_out, int out_size, void* d_ws, size_t ws_size,
                              hipStream_t stream) {
  const float* x      = (const float*)d_in[0];
  const int*   ei     = (const int*)d_in[1];
  const float* W1     = (const float*)d_in[2];
  const float* att_s1 = (const float*)d_in[3];
  const float* att_d1 = (const float*)d_in[4];
  const float* b1     = (const float*)d_in[5];
  const float* W2     = (const float*)d_in[6];
  const float* att_s2 = (const float*)d_in[7];
  const float* att_d2 = (const float*)d_in[8];
  const float* b2     = (const float*)d_in[9];
  float* out = (float*)d_out;

  const int  Nn   = in_sizes[0] / 256;
  const int  Ee   = in_sizes[1] / 2;
  const long Etot = (long)Ee + Nn;
  const int* e_src = ei;
  const int* e_dst = ei + Ee;

  // ---- workspace layout (region reuse: xh->h2h, h1->g2+agg2) -------------
  char* p = (char*)d_ws;
  _Float16* xh    = (_Float16*)ws_alloc(p, (size_t)Nn * 256 * 2);   // later h2h
  _Float16* W1t   = (_Float16*)ws_alloc(p, 256 * 256 * 2);          // transposed
  _Float16* W2t   = (_Float16*)ws_alloc(p, 48 * 256 * 2);           // transposed+pad
  float*    h1    = (float*)   ws_alloc(p, (size_t)Nn * 256 * 4);   // later g2/agg2
  float*    agg1  = (float*)   ws_alloc(p, (size_t)Nn * 256 * 4);
  float*    as1   = (float*)   ws_alloc(p, (size_t)Nn * 8 * 4);
  float*    ad1   = (float*)   ws_alloc(p, (size_t)Nn * 8 * 4);
  unsigned* m1    = (unsigned*)ws_alloc(p, (size_t)Nn * 8 * 4);
  float*    den1  = (float*)   ws_alloc(p, (size_t)Nn * 8 * 4);
  float*    as2   = (float*)   ws_alloc(p, (size_t)Nn * 4);
  float*    ad2   = (float*)   ws_alloc(p, (size_t)Nn * 4);
  unsigned* m2    = (unsigned*)ws_alloc(p, (size_t)Nn * 4);
  float*    den2  = (float*)   ws_alloc(p, (size_t)Nn * 4);
  _Float16* h2h   = xh;                       // layer-1 done before layer-2 GEMM
  float*    g2    = h1;                       // [Nn,48], reuses h1 region
  float*    agg2  = h1 + (size_t)Nn * 48;     // [Nn,40]

  const int BT = 256;
  auto blocks = [](long n, int bt) { return (unsigned)((n + bt - 1) / bt); };

  // ---- fp16 conversions (weights pre-transposed for vectorized staging) --
  gat_cvt_f16<<<blocks((long)Nn * 256, BT), BT, 0, stream>>>(x, xh, (long)Nn * 256);
  gat_cvt_w1t<<<blocks(256 * 256, BT), BT, 0, stream>>>(W1, W1t);
  gat_cvt_w2t<<<blocks(48 * 256, BT), BT, 0, stream>>>(W2, W2t);

  // ---- layer 1: GEMM + alpha ---------------------------------------------
  const unsigned gx = (unsigned)((Nn + 127) / 128);
  gat_wmma_gemm<4><<<dim3(gx, 4), 256, 0, stream>>>(xh, W1t, h1, Nn, 256, 256, 256);
  gat_alpha_l1<<<blocks((long)Nn * 8, BT), BT, 0, stream>>>(h1, att_s1, att_d1, as1, ad1, Nn);

  // ---- layer 1: edge softmax + aggregate ---------------------------------
  gat_fill_u32<<<blocks((long)Nn * 8, BT), BT, 0, stream>>>(m1, NEG_INF_ORD, (long)Nn * 8);
  gat_fill_u32<<<blocks((long)Nn * 8, BT), BT, 0, stream>>>((unsigned*)den1, 0u, (long)Nn * 8);
  gat_fill_u32<<<blocks((long)Nn * 256, BT), BT, 0, stream>>>((unsigned*)agg1, 0u, (long)Nn * 256);
  gat_edge_max<8><<<blocks(Etot * 8, BT), BT, 0, stream>>>(e_src, e_dst, Ee, Nn, as1, ad1, m1);
  gat_edge_sum<8><<<blocks(Etot * 8, BT), BT, 0, stream>>>(e_src, e_dst, Ee, Nn, as1, ad1, m1, den1);
  gat_edge_agg_h8<<<blocks(Etot * 32, BT), BT, 0, stream>>>(e_src, e_dst, Ee, Nn,
                                                            as1, ad1, m1, den1, h1, agg1);

  // ---- ELU + bias -> f16 features for layer 2 -----------------------------
  gat_elu_f16<<<blocks((long)Nn * 256, BT), BT, 0, stream>>>(agg1, b1, h2h, (long)Nn * 256);

  // ---- layer 2: GEMM (40 cols padded to 48) + alpha -----------------------
  gat_wmma_gemm<3><<<dim3(gx, 1), 256, 0, stream>>>(h2h, W2t, g2, Nn, 256, 48, 48);
  gat_alpha_l2<<<blocks(Nn, BT), BT, 0, stream>>>(g2, att_s2, att_d2, as2, ad2, Nn);

  // ---- layer 2: edge softmax + aggregate ---------------------------------
  gat_fill_u32<<<blocks(Nn, BT), BT, 0, stream>>>(m2, NEG_INF_ORD, Nn);
  gat_fill_u32<<<blocks(Nn, BT), BT, 0, stream>>>((unsigned*)den2, 0u, Nn);
  gat_fill_u32<<<blocks((long)Nn * 40, BT), BT, 0, stream>>>((unsigned*)agg2, 0u, (long)Nn * 40);
  gat_edge_max<1><<<blocks(Etot, BT), BT, 0, stream>>>(e_src, e_dst, Ee, Nn, as2, ad2, m2);
  gat_edge_sum<1><<<blocks(Etot, BT), BT, 0, stream>>>(e_src, e_dst, Ee, Nn, as2, ad2, m2, den2);
  gat_edge_agg_h1<<<blocks(Etot * 32, BT), BT, 0, stream>>>(e_src, e_dst, Ee, Nn,
                                                            as2, ad2, m2, den2, g2, agg2);

  // ---- output -------------------------------------------------------------
  gat_final_bias<<<blocks((long)Nn * 40, BT), BT, 0, stream>>>(agg2, b2, out, (long)Nn * 40);
}
